// ChunkBasedHST_85761906967027
// MI455X (gfx1250) — compile-verified, matrix operand order checked by reading, not run on hardware
//
#include <hip/hip_runtime.h>
#include <hip/hip_bf16.h>
#include <cmath>

// ---------------- model constants ----------------
#define DMODEL   256
#define DFF      1024
#define CHUNKSZ  128
#define NCH      16
#define TOK      2048
#define VOCAB    32000
#define NHEAD    8
#define HEADD    32
#define NSPINE   4
#define LN_EPS   1e-5f

typedef __attribute__((ext_vector_type(16))) _Float16 v16h;
typedef __attribute__((ext_vector_type(8)))  float    v8f;

// K-index mapping inside a 16x32 (f16) WMMA A/B fragment (CDNA5 ISA):
// lanes 0-15: halves 0-7 -> K0-7,  halves 8-15 -> K16-23
// lanes16-31: halves 0-7 -> K8-15, halves 8-15 -> K24-31
__device__ __forceinline__ int frag_k(int lane, int h) {
  int g = (lane >> 4) & 1;
  return (h < 8) ? (g * 8 + h) : (16 + g * 8 + (h - 8));
}

// ---------------- WMMA GEMM: C = act(A[MxK] @ B[KxN] + bias) ----------------
// Row-major, tight leading dims. Block tile 128x64, 8 waves of 32x32.
// B tile is staged through LDS (transposed, padded) so fragment reads are
// contiguous ds_load_b128 runs; interior A fragments vectorize to b128 loads.
#define BT_STRIDE 36   // 32 + 4 pad floats: 16 lanes -> 16 distinct banks, 16B-aligned runs

template<bool RELU, bool OUT_HALF>
__global__ __launch_bounds__(256)
void gemm_kernel(const float* __restrict__ A, const float* __restrict__ B,
                 const float* __restrict__ bias,
                 float* __restrict__ Cf, _Float16* __restrict__ Ch,
                 int M, int N, int K)
{
  __shared__ float bt[64 * BT_STRIDE];   // B tile transposed: bt[n][k]
  const int tid  = threadIdx.x;
  const int lane = tid & 31;
  const int wave = tid >> 5;
  const int wm = wave & 3;               // 4 waves along M
  const int wn = wave >> 2;              // 2 waves along N
  const int r0 = blockIdx.x * 128 + wm * 32;
  const int c0 = blockIdx.y * 64;        // block column base
  const int cw = wn * 32;                // wave column offset inside tile
  const int rr = lane & 15;
  const int g  = (lane >> 4) & 1;
  const bool a_full = (r0 + 32 <= M);

  v8f acc[2][2] = {};

  for (int k0 = 0; k0 < K; k0 += 32) {
    // ---- stage B tile: coalesced along N, transposed into LDS ----
#pragma unroll
    for (int i = 0; i < 8; ++i) {
      const int e  = tid + i * 256;      // 2048 elements = 32(K) x 64(N)
      const int kk = e >> 6;             // 0..31
      const int n  = e & 63;             // 0..63
      int col = c0 + n; if (col >= N) col = N - 1;   // clamp (stays in-bounds; masked at store)
      bt[n * BT_STRIDE + kk] = B[(size_t)(k0 + kk) * N + col];
    }
    __syncthreads();

    // ---- A fragments ----
    v16h af[2];
    if (a_full) {
      const float* ar0 = A + (size_t)(r0 + rr)      * K + k0 + g * 8;
      const float* ar1 = A + (size_t)(r0 + 16 + rr) * K + k0 + g * 8;
      if (k0 + 32 < K) __builtin_prefetch(ar0 + 32, 0, 1);
#pragma unroll
      for (int t = 0; t < 8; ++t) {
        af[0][t]     = (_Float16)ar0[t];
        af[0][t + 8] = (_Float16)ar0[16 + t];
        af[1][t]     = (_Float16)ar1[t];
        af[1][t + 8] = (_Float16)ar1[16 + t];
      }
    } else {
#pragma unroll
      for (int h = 0; h < 16; ++h) {
        const int kk = k0 + frag_k(lane, h);
        const int ra = r0 + rr, rb = r0 + 16 + rr;
        af[0][h] = (_Float16)((ra < M) ? A[(size_t)ra * K + kk] : 0.f);
        af[1][h] = (_Float16)((rb < M) ? A[(size_t)rb * K + kk] : 0.f);
      }
    }

    // ---- B fragments from LDS: two contiguous 8-float runs per lane ----
    v16h bf[2];
#pragma unroll
    for (int j = 0; j < 2; ++j) {
      const float* br = bt + (cw + j * 16 + rr) * BT_STRIDE + g * 8;
#pragma unroll
      for (int t = 0; t < 8; ++t) {
        bf[j][t]     = (_Float16)br[t];
        bf[j][t + 8] = (_Float16)br[16 + t];
      }
    }

#pragma unroll
    for (int i = 0; i < 2; ++i)
#pragma unroll
      for (int j = 0; j < 2; ++j)
        acc[i][j] = __builtin_amdgcn_wmma_f32_16x16x32_f16(
            false, af[i], false, bf[j], (short)0, acc[i][j], false, false);

    __syncthreads();   // WAR: protect bt before next stage
  }

#pragma unroll
  for (int i = 0; i < 2; ++i) {
#pragma unroll
    for (int j = 0; j < 2; ++j) {
#pragma unroll
      for (int e = 0; e < 8; ++e) {
        const int m = r0 + i * 16 + e + g * 8;
        const int n = c0 + cw + j * 16 + rr;
        if (m < M && n < N) {
          float v = acc[i][j][e] + (bias ? bias[n] : 0.f);
          if (RELU) v = v > 0.f ? v : 0.f;
          if (OUT_HALF) Ch[(size_t)m * N + n] = (_Float16)v;
          else          Cf[(size_t)m * N + n] = v;
        }
      }
    }
  }
}

// ---------------- flash attention: one wave per 16-query tile ----------------
// Q,K,V: f16 [T, DMODEL] with per-head 32-col slices. O: f32 [T, DMODEL].
__global__ __launch_bounds__(32)
void attn_kernel(const _Float16* __restrict__ Q, const _Float16* __restrict__ K,
                 const _Float16* __restrict__ V, float* __restrict__ O,
                 int Tk, int chunkQ, int chunkK, float scale)
{
  __shared__ _Float16 pl[16 * 32];
  const int lane  = threadIdx.x;
  const int head  = blockIdx.y;
  const int chunk = blockIdx.z;
  const int q0    = blockIdx.x * 16;
  const _Float16* qb = Q + (size_t)chunk * chunkQ * DMODEL + head * HEADD;
  const _Float16* kb = K + (size_t)chunk * chunkK * DMODEL + head * HEADD;
  const _Float16* vb = V + (size_t)chunk * chunkK * DMODEL + head * HEADD;
  float*          ob = O + (size_t)chunk * chunkQ * DMODEL + head * HEADD;
  const int rr = lane & 15;
  const int g  = (lane >> 4) & 1;

  // Q fragment: rows = 16 queries, K-dim = 32 head dims
  v16h qf;
#pragma unroll
  for (int h = 0; h < 16; ++h)
    qf[h] = qb[(size_t)(q0 + rr) * DMODEL + frag_k(lane, h)];

  v8f o0 = {}, o1 = {};
  float mrow[8], lrow[8];
#pragma unroll
  for (int j = 0; j < 8; ++j) { mrow[j] = -1e30f; lrow[j] = 0.f; }

  for (int kt = 0; kt < Tk; kt += 32) {
    if (kt + 32 < Tk)
      __builtin_prefetch(kb + (size_t)(kt + 32 + rr) * DMODEL, 0, 1);

    // S = Q @ K^T for 32 keys: B[dim, key] = K[key, dim]
    v16h kf0, kf1;
#pragma unroll
    for (int h = 0; h < 16; ++h) {
      const int dd = frag_k(lane, h);
      kf0[h] = kb[(size_t)(kt +      rr) * DMODEL + dd];
      kf1[h] = kb[(size_t)(kt + 16 + rr) * DMODEL + dd];
    }
    v8f s0 = {}, s1 = {};
    s0 = __builtin_amdgcn_wmma_f32_16x16x32_f16(false, qf, false, kf0, (short)0, s0, false, false);
    s1 = __builtin_amdgcn_wmma_f32_16x16x32_f16(false, qf, false, kf1, (short)0, s1, false, false);

    // online softmax: rows live across 16-lane groups of the C layout
    float p0[8], p1[8];
#pragma unroll
    for (int j = 0; j < 8; ++j) {
      float v0 = s0[j] * scale, v1 = s1[j] * scale;
      float t = fmaxf(v0, v1);
#pragma unroll
      for (int msk = 1; msk < 16; msk <<= 1) t = fmaxf(t, __shfl_xor(t, msk, 32));
      const float mnew  = fmaxf(mrow[j], t);
      const float alpha = __expf(mrow[j] - mnew);
      mrow[j] = mnew;
      p0[j] = __expf(v0 - mnew);
      p1[j] = __expf(v1 - mnew);
      float rs = p0[j] + p1[j];
#pragma unroll
      for (int msk = 1; msk < 16; msk <<= 1) rs += __shfl_xor(rs, msk, 32);
      lrow[j] = lrow[j] * alpha + rs;
      o0[j] *= alpha;
      o1[j] *= alpha;
    }

    // C-layout P tile -> LDS -> reload in A-fragment layout
#pragma unroll
    for (int j = 0; j < 8; ++j) {
      const int row = j + g * 8;
      pl[row * 32 + rr]      = (_Float16)p0[j];
      pl[row * 32 + 16 + rr] = (_Float16)p1[j];
    }
    asm volatile("s_wait_dscnt 0" ::: "memory");
    v16h pf;
#pragma unroll
    for (int h = 0; h < 16; ++h)
      pf[h] = pl[rr * 32 + frag_k(lane, h)];

    // O += P @ V : B[key, dim] = V[kt+key, dim]
    v16h vf0, vf1;
#pragma unroll
    for (int h = 0; h < 16; ++h) {
      const int kk = frag_k(lane, h);
      vf0[h] = vb[(size_t)(kt + kk) * DMODEL + rr];
      vf1[h] = vb[(size_t)(kt + kk) * DMODEL + 16 + rr];
    }
    o0 = __builtin_amdgcn_wmma_f32_16x16x32_f16(false, pf, false, vf0, (short)0, o0, false, false);
    o1 = __builtin_amdgcn_wmma_f32_16x16x32_f16(false, pf, false, vf1, (short)0, o1, false, false);
  }

#pragma unroll
  for (int j = 0; j < 8; ++j) {
    const int row = q0 + j + g * 8;
    const float inv = 1.f / lrow[j];
    ob[(size_t)row * DMODEL + rr]      = o0[j] * inv;
    ob[(size_t)row * DMODEL + 16 + rr] = o1[j] * inv;
  }
}

// ---------------- fused (residual-add +) LayerNorm, one row / block ----------------
template<bool RESID>
__global__ __launch_bounds__(256)
void add_ln_kernel(const float* __restrict__ x, const float* __restrict__ r,
                   const float* __restrict__ gam, const float* __restrict__ bet,
                   float* __restrict__ out)
{
  __shared__ float red[DMODEL];
  const int row = blockIdx.x, d = threadIdx.x;
  float v = x[(size_t)row * DMODEL + d];
  if (RESID) v += r[(size_t)row * DMODEL + d];
  red[d] = v; __syncthreads();
  for (int s = DMODEL / 2; s > 0; s >>= 1) { if (d < s) red[d] += red[d + s]; __syncthreads(); }
  const float mean = red[0] / DMODEL; __syncthreads();
  const float c = v - mean;
  red[d] = c * c; __syncthreads();
  for (int s = DMODEL / 2; s > 0; s >>= 1) { if (d < s) red[d] += red[d + s]; __syncthreads(); }
  const float var = red[0] / DMODEL;
  out[(size_t)row * DMODEL + d] = c * rsqrtf(var + LN_EPS) * gam[d] + bet[d];
}

// ---------------- glue kernels ----------------
__global__ void embed_kernel(const int* __restrict__ ids, const float* __restrict__ tok,
                             const float* __restrict__ pos, float* __restrict__ x)
{
  const int t = blockIdx.x, d = threadIdx.x;
  x[(size_t)t * DMODEL + d] = tok[(size_t)ids[t] * DMODEL + d] + pos[(size_t)t * DMODEL + d];
}

__global__ void pool_kernel(const float* __restrict__ h, float* __restrict__ pooled)
{
  const int c = blockIdx.x, d = threadIdx.x;
  float s = 0.f;
  for (int t = 0; t < CHUNKSZ; ++t) s += h[(size_t)(c * CHUNKSZ + t) * DMODEL + d];
  pooled[(size_t)c * DMODEL + d] = s * (1.f / CHUNKSZ);
}

__global__ void spine_agg_kernel(const float* __restrict__ ce, float* __restrict__ agg)
{
  const int spine[NSPINE] = {0, 2, 4, 12};
  const int p = spine[blockIdx.x], d = threadIdx.x;
  float s = 0.f; int n = 0;
  if (p > 0)       { s += ce[(size_t)(p - 1) * DMODEL + d]; ++n; }
  s += ce[(size_t)p * DMODEL + d]; ++n;
  if (p < NCH - 1) { s += ce[(size_t)(p + 1) * DMODEL + d]; ++n; }
  agg[(size_t)blockIdx.x * DMODEL + d] = s / (float)n;
}

__global__ void concat_kernel(const float* __restrict__ ce, const float* __restrict__ a,
                              float* __restrict__ comb)
{
  const int spine[NSPINE] = {0, 2, 4, 12};
  const int pi = blockIdx.x, d = threadIdx.x;
  comb[(size_t)pi * 2 * DMODEL + d]          = ce[(size_t)spine[pi] * DMODEL + d];
  comb[(size_t)pi * 2 * DMODEL + DMODEL + d] = a[(size_t)pi * DMODEL + d];
}

__global__ void scatter_kernel(const float* __restrict__ ce, const float* __restrict__ inj,
                               float* __restrict__ hlat)
{
  const int c = blockIdx.x, d = threadIdx.x;
  int pi = -1;
  if (c == 0) pi = 0; else if (c == 2) pi = 1; else if (c == 4) pi = 2; else if (c == 12) pi = 3;
  hlat[(size_t)c * DMODEL + d] =
      (pi >= 0) ? inj[(size_t)pi * DMODEL + d] : ce[(size_t)c * DMODEL + d];
}

// ---------------- host-side helpers ----------------
static void gemm_f32(hipStream_t s, const float* A, const float* B, const float* bias,
                     float* C, int M, int N, int K, bool relu)
{
  dim3 grid((M + 127) / 128, (N + 63) / 64);
  if (relu) gemm_kernel<true, false><<<grid, 256, 0, s>>>(A, B, bias, C, nullptr, M, N, K);
  else      gemm_kernel<false, false><<<grid, 256, 0, s>>>(A, B, bias, C, nullptr, M, N, K);
}
static void gemm_f16(hipStream_t s, const float* A, const float* B, const float* bias,
                     _Float16* C, int M, int N, int K)
{
  dim3 grid((M + 127) / 128, (N + 63) / 64);
  gemm_kernel<false, true><<<grid, 256, 0, s>>>(A, B, bias, nullptr, C, M, N, K);
}

struct Bufs {
  float *X, *TMP, *ATT, *H1, *MEM, *POOLED, *CE, *LA, *LT, *LT2, *LH1, *COMB, *INJ, *HLAT;
  _Float16 *QH, *KH, *VH;
};

// encoder layer over T rows; attention confined to chunks of chunkT
static void enc_layer(hipStream_t s, const Bufs& w, float* X, int T, int nchunks, int chunkT,
                      void* const* din, int base)
{
  auto F = [&](int i) { return (const float*)din[base + i]; };
  const float scale = 1.f / sqrtf((float)HEADD);
  gemm_f16(s, X, F(0), F(1), w.QH, T, DMODEL, DMODEL); // Wq,bq
  gemm_f16(s, X, F(2), F(3), w.KH, T, DMODEL, DMODEL); // Wk,bk
  gemm_f16(s, X, F(4), F(5), w.VH, T, DMODEL, DMODEL); // Wv,bv
  dim3 ag(chunkT / 16, NHEAD, nchunks);
  attn_kernel<<<ag, 32, 0, s>>>(w.QH, w.KH, w.VH, w.ATT, chunkT, chunkT, chunkT, scale);
  gemm_f32(s, w.ATT, F(6), F(7), w.TMP, T, DMODEL, DMODEL, false); // Wo,bo
  add_ln_kernel<true><<<T, DMODEL, 0, s>>>(w.TMP, X, F(8), F(9), X); // ln1
  gemm_f32(s, X, F(10), F(11), w.H1, T, DFF, DMODEL, true);  // W1,b1 relu
  gemm_f32(s, w.H1, F(12), F(13), w.TMP, T, DMODEL, DFF, false); // W2,b2
  add_ln_kernel<true><<<T, DMODEL, 0, s>>>(w.TMP, X, F(14), F(15), X); // ln2
}

// decoder layer, full-sequence self+cross attention
static void dec_layer(hipStream_t s, const Bufs& w, float* X, const float* MEM,
                      void* const* din, int base)
{
  auto F = [&](int i) { return (const float*)din[base + i]; };
  const float scale = 1.f / sqrtf((float)HEADD);
  dim3 ag(TOK / 16, NHEAD, 1);
  // self-attention (sa = base+0..7)
  gemm_f16(s, X, F(0), F(1), w.QH, TOK, DMODEL, DMODEL);
  gemm_f16(s, X, F(2), F(3), w.KH, TOK, DMODEL, DMODEL);
  gemm_f16(s, X, F(4), F(5), w.VH, TOK, DMODEL, DMODEL);
  attn_kernel<<<ag, 32, 0, s>>>(w.QH, w.KH, w.VH, w.ATT, TOK, TOK, TOK, scale);
  gemm_f32(s, w.ATT, F(6), F(7), w.TMP, TOK, DMODEL, DMODEL, false);
  add_ln_kernel<true><<<TOK, DMODEL, 0, s>>>(w.TMP, X, F(16), F(17), X); // ln1
  // cross-attention (ca = base+8..15), memory = MEM
  gemm_f16(s, X,   F(8),  F(9),  w.QH, TOK, DMODEL, DMODEL);
  gemm_f16(s, MEM, F(10), F(11), w.KH, TOK, DMODEL, DMODEL);
  gemm_f16(s, MEM, F(12), F(13), w.VH, TOK, DMODEL, DMODEL);
  attn_kernel<<<ag, 32, 0, s>>>(w.QH, w.KH, w.VH, w.ATT, TOK, TOK, TOK, scale);
  gemm_f32(s, w.ATT, F(14), F(15), w.TMP, TOK, DMODEL, DMODEL, false);
  add_ln_kernel<true><<<TOK, DMODEL, 0, s>>>(w.TMP, X, F(18), F(19), X); // ln2
  // FFN
  gemm_f32(s, X, F(20), F(21), w.H1, TOK, DFF, DMODEL, true);
  gemm_f32(s, w.H1, F(22), F(23), w.TMP, TOK, DMODEL, DFF, false);
  add_ln_kernel<true><<<TOK, DMODEL, 0, s>>>(w.TMP, X, F(24), F(25), X); // ln3
}

// lattice encoder layer with seq_len==1: mha(x,x) == (x@Wv+bv)@Wo+bo
static void lattice_layer(hipStream_t s, const Bufs& w, void* const* din, int base)
{
  auto F = [&](int i) { return (const float*)din[base + i]; };
  gemm_f32(s, w.LA, F(4), F(5), w.LT, NSPINE, DMODEL, DMODEL, false); // Wv,bv
  gemm_f32(s, w.LT, F(6), F(7), w.LT2, NSPINE, DMODEL, DMODEL, false); // Wo,bo
  add_ln_kernel<true><<<NSPINE, DMODEL, 0, s>>>(w.LT2, w.LA, F(8), F(9), w.LA);
  gemm_f32(s, w.LA, F(10), F(11), w.LH1, NSPINE, DFF, DMODEL, true);
  gemm_f32(s, w.LH1, F(12), F(13), w.LT, NSPINE, DMODEL, DFF, false);
  add_ln_kernel<true><<<NSPINE, DMODEL, 0, s>>>(w.LT, w.LA, F(14), F(15), w.LA);
}

extern "C" void kernel_launch(void* const* d_in, const int* in_sizes, int n_in,
                              void* d_out, int out_size, void* d_ws, size_t ws_size,
                              hipStream_t stream)
{
  (void)in_sizes; (void)n_in; (void)out_size; (void)ws_size;
  // ---- workspace layout ----
  char* p = (char*)d_ws;
  auto take = [&](size_t bytes) { char* r = p; p += (bytes + 255) & ~(size_t)255; return r; };
  Bufs w;
  w.X      = (float*)take((size_t)TOK * DMODEL * 4);
  w.TMP    = (float*)take((size_t)TOK * DMODEL * 4);
  w.ATT    = (float*)take((size_t)TOK * DMODEL * 4);
  w.H1     = (float*)take((size_t)TOK * DFF * 4);
  w.MEM    = (float*)take((size_t)TOK * DMODEL * 4);
  w.QH     = (_Float16*)take((size_t)TOK * DMODEL * 2);
  w.KH     = (_Float16*)take((size_t)TOK * DMODEL * 2);
  w.VH     = (_Float16*)take((size_t)TOK * DMODEL * 2);
  w.POOLED = (float*)take((size_t)NCH * DMODEL * 4);
  w.CE     = (float*)take((size_t)NCH * DMODEL * 4);
  w.LA     = (float*)take((size_t)NSPINE * DMODEL * 4);
  w.LT     = (float*)take((size_t)NSPINE * DMODEL * 4);
  w.LT2    = (float*)take((size_t)NSPINE * DMODEL * 4);
  w.LH1    = (float*)take((size_t)NSPINE * DFF * 4);
  w.COMB   = (float*)take((size_t)NSPINE * 2 * DMODEL * 4);
  w.INJ    = (float*)take((size_t)NSPINE * DMODEL * 4);
  w.HLAT   = (float*)take((size_t)NCH * DMODEL * 4);

  // ---- input index map (setup_inputs() insertion order) ----
  const int* ids      = (const int*)d_in[0];
  const float* tok    = (const float*)d_in[1];
  const float* pos    = (const float*)d_in[2];
  const int ENC0 = 3, ENC1 = 19;             // 16 tensors per encoder layer
  const int POOL_W = 35, POOL_B = 36, POOL_G = 37, POOL_BETA = 38;
  const int LAT0 = 39;                       // 4 layers x 16 tensors
  const int INJ_W = 103, INJ_B = 104;
  const int EXP_W = 105, EXP_B = 106;
  const int DEC0 = 107, DEC1 = 133;          // 26 tensors per decoder layer
  const int LM_W = 159, LM_B = 160;
  float* out = (float*)d_out;

  // 1) embedding
  embed_kernel<<<TOK, DMODEL, 0, stream>>>(ids, tok, pos, w.X);

  // 2) chunk encoder (attention confined to 16 chunks of 128)
  enc_layer(stream, w, w.X, TOK, NCH, CHUNKSZ, d_in, ENC0);
  enc_layer(stream, w, w.X, TOK, NCH, CHUNKSZ, d_in, ENC1);

  // 3) pool -> project -> LN  => chunk_emb
  pool_kernel<<<NCH, DMODEL, 0, stream>>>(w.X, w.POOLED);
  gemm_f32(stream, w.POOLED, (const float*)d_in[POOL_W], (const float*)d_in[POOL_B],
           w.TMP, NCH, DMODEL, DMODEL, false);
  add_ln_kernel<false><<<NCH, DMODEL, 0, stream>>>(
      w.TMP, nullptr, (const float*)d_in[POOL_G], (const float*)d_in[POOL_BETA], w.CE);

  // 4) lattice spine: neighborhood means, then 4 seq_len==1 encoder layers
  spine_agg_kernel<<<NSPINE, DMODEL, 0, stream>>>(w.CE, w.LA);
  for (int l = 0; l < 4; ++l)
    lattice_layer(stream, w, d_in, LAT0 + 16 * l);

  // 5) inject: concat(chunk_emb[spine], a) @ inj_W + inj_b, scatter into h_lattice
  concat_kernel<<<NSPINE, DMODEL, 0, stream>>>(w.CE, w.LA, w.COMB);
  gemm_f32(stream, w.COMB, (const float*)d_in[INJ_W], (const float*)d_in[INJ_B],
           w.INJ, NSPINE, DMODEL, 2 * DMODEL, false);
  scatter_kernel<<<NCH, DMODEL, 0, stream>>>(w.CE, w.INJ, w.HLAT);

  // 6) expand: [16,256] @ [256, 32768] -> memory [2048,256]
  gemm_f32(stream, w.HLAT, (const float*)d_in[EXP_W], (const float*)d_in[EXP_B],
           w.MEM, NCH, DMODEL * CHUNKSZ, DMODEL, false);

  // 7) decoder: tgt starts as expanded memory
  hipMemcpyAsync(w.X, w.MEM, (size_t)TOK * DMODEL * 4, hipMemcpyDeviceToDevice, stream);
  dec_layer(stream, w, w.X, w.MEM, d_in, DEC0);
  dec_layer(stream, w, w.X, w.MEM, d_in, DEC1);

  // 8) LM head -> d_out  [2048, 32000] f32
  gemm_f32(stream, w.X, (const float*)d_in[LM_W], (const float*)d_in[LM_B],
           out, TOK, VOCAB, DMODEL, false);
}
